// SimpleSNN_14740327760336
// MI455X (gfx1250) — matrix operand order repaired and checked by reading, not compiled, not run on hardware
//
#include <hip/hip_runtime.h>
#include <hip/hip_bf16.h>

// ---------------------------------------------------------------------------
// SimpleSNN on MI455X (gfx1250, wave32, WMMA + async-to-LDS staging)
//   emb gather -> FC1 (WMMA f16/fp32 acc) -> membrane scan -> FC2 (WMMA) -> logits
// FC2 dominates (134 GFLOP); W2 (65MB f16) lives in the 192MB L2. B tiles are
// shared by all 4 waves of a block -> stage them in LDS via
// global_load_async_to_lds_b128 (ASYNCcnt) with double buffering.
// ---------------------------------------------------------------------------

typedef __attribute__((ext_vector_type(16))) _Float16 v16h;
typedef __attribute__((ext_vector_type(8)))  _Float16 v8h;
typedef __attribute__((ext_vector_type(8)))  float    v8f;

#define VOCAB   32000
#define DMODEL  512
#define HIDDEN  1024
#define BB      4
#define TT      512
#define MTOT    (BB * TT)          // 2048
#define DECAY_F 0.60653065971263342f   // exp(-1/2)

// LDS B tile: [2 buffers][64 N-rows][40 halves] (32 K-halves + 8 pad)
// row stride 80 B => 20-bank stride, conflict-free for 16-lane b128 reads,
// and every 16B chunk stays 16B-aligned.
#define LDSB_ROWH 40
#define LDSB_ROWB 80
#define LDSB_BUFB (64 * LDSB_ROWB)    // 5120 B per buffer

__device__ __forceinline__ void async_b128_to_lds(unsigned lds_off, const void* gptr) {
    asm volatile("global_load_async_to_lds_b128 %0, %1, off"
                 :: "v"(lds_off), "v"((unsigned long long)(uintptr_t)gptr)
                 : "memory");
}

// ---------------- generic f32 -> f16 cast ----------------
__global__ void cast_f32_to_f16(const float* __restrict__ src,
                                _Float16* __restrict__ dst, size_t n) {
    size_t i = (size_t)blockIdx.x * blockDim.x + threadIdx.x;
    if (i < n) dst[i] = (_Float16)src[i];
}

// ---------------- embedding gather + cast ----------------
__global__ void gather_cast(const int* __restrict__ ids,
                            const float* __restrict__ emb,
                            _Float16* __restrict__ xh) {
    size_t i = (size_t)blockIdx.x * blockDim.x + threadIdx.x;   // m*D + d
    if (i >= (size_t)MTOT * DMODEL) return;
    int m = (int)(i / DMODEL);
    int d = (int)(i % DMODEL);
    xh[i] = (_Float16)emb[(size_t)ids[m] * DMODEL + d];
}

// ---------------- membrane potential scan (sequential in T) ----------------
__global__ void snn_scan(const float* __restrict__ u,
                         const float* __restrict__ thr,
                         _Float16* __restrict__ spikes) {
    int idx = blockIdx.x * blockDim.x + threadIdx.x;   // b*H + h
    if (idx >= BB * HIDDEN) return;
    int b = idx / HIDDEN;
    int h = idx % HIDDEN;
    const float th = thr[h];
    const float* up = u + (size_t)b * TT * HIDDEN + h;
    _Float16*   sp = spikes + (size_t)b * TT * HIDDEN + h;
    float mem = 0.0f;
    for (int t = 0; t < TT; ++t) {
        mem = mem * DECAY_F + up[(size_t)t * HIDDEN];
        float s = (mem >= th) ? 1.0f : 0.0f;
        sp[(size_t)t * HIDDEN] = (_Float16)s;
        mem = (s != 0.0f) ? 0.0f : mem;
    }
}

// ---------------- WMMA GEMM: C[M,N] = A[M,K] * Bw[N,K]^T + bias ----------------
// Block = 128 threads = 4 waves. Waves split M (32 rows each); all share the
// block's 64-wide N tile, whose 64x32 B slab is staged in LDS via async DMA.
__global__ __launch_bounds__(128) void gemm_f16_wmma(
    const _Float16* __restrict__ A,
    const _Float16* __restrict__ Bw,
    const float* __restrict__ bias,
    float* __restrict__ C,
    int M, int N, int K)
{
    __shared__ _Float16 ldsB[2][64][LDSB_ROWH];

    const int lane  = threadIdx.x & 31;
    const int wave  = threadIdx.x >> 5;          // 0..3
    const int tileN = blockIdx.x * 64;
    const int tileM = blockIdx.y * 128 + wave * 32;

    const int lrow  = lane & 15;                  // M-row / N-col within tile
    const int lhalf = lane >> 4;                  // 0 or 1
    const int akb   = lhalf * 8;                  // A lane K-base
    const int bkb   = lhalf * 8;                  // B lane K-base

    const unsigned ldsBase = (unsigned)(uintptr_t)(&ldsB[0][0][0]);

    // per-thread async-copy addressing: 256 16B chunks = 128 threads x 2
    const int c0   = (int)threadIdx.x * 2;        // chunk ids c0, c0+1
    const int row0 = c0 >> 2,       sub0 = c0 & 3;
    const int row1 = (c0 + 1) >> 2, sub1 = (c0 + 1) & 3;
    const _Float16* bg0 = Bw + (size_t)(tileN + row0) * K + sub0 * 8;
    const _Float16* bg1 = Bw + (size_t)(tileN + row1) * K + sub1 * 8;
    const unsigned  bl0 = ldsBase + (unsigned)(row0 * LDSB_ROWB + sub0 * 16);
    const unsigned  bl1 = ldsBase + (unsigned)(row1 * LDSB_ROWB + sub1 * 16);

    v8f acc[2][4];
#pragma unroll
    for (int mt = 0; mt < 2; ++mt)
#pragma unroll
        for (int nt = 0; nt < 4; ++nt)
            acc[mt][nt] = (v8f){0.f,0.f,0.f,0.f,0.f,0.f,0.f,0.f};

    const _Float16* aBase[2];
    aBase[0] = A + (size_t)(tileM +      lrow) * K + akb;
    aBase[1] = A + (size_t)(tileM + 16 + lrow) * K + akb;

    const int nsteps = K >> 5;

    // preload buffer 0
    async_b128_to_lds(bl0, bg0);
    async_b128_to_lds(bl1, bg1);

    for (int ks = 0; ks < nsteps; ++ks) {
        const int k0  = ks << 5;
        const int buf = ks & 1;

        if (ks + 1 < nsteps) {
            const int kn = k0 + 32;
            const unsigned bofs = (unsigned)((buf ^ 1) * LDSB_BUFB);
            async_b128_to_lds(bl0 + bofs, bg0 + kn);
            async_b128_to_lds(bl1 + bofs, bg1 + kn);
            asm volatile("s_wait_asynccnt 0x2" ::: "memory");
        } else {
            asm volatile("s_wait_asynccnt 0x0" ::: "memory");
        }
        __syncthreads();   // current buffer visible to all waves

        v16h af[2];
#pragma unroll
        for (int mt = 0; mt < 2; ++mt) {
            v8h lo = *(const v8h*)(aBase[mt] + k0);        // K = k0+akb   .. +7
            v8h hi = *(const v8h*)(aBase[mt] + k0 + 16);   // K = k0+akb+16.. +23
            af[mt] = __builtin_shufflevector(lo, hi,
                        0,1,2,3,4,5,6,7,8,9,10,11,12,13,14,15);
        }
        v16h bf[4];
#pragma unroll
        for (int nt = 0; nt < 4; ++nt) {
            const _Float16* bp = &ldsB[buf][nt * 16 + lrow][bkb];
            v8h blo = *(const v8h*)(bp);
            v8h bhi = *(const v8h*)(bp + 16);
            bf[nt] = __builtin_shufflevector(blo, bhi,
                        0,1,2,3,4,5,6,7,8,9,10,11,12,13,14,15);
        }

#pragma unroll
        for (int mt = 0; mt < 2; ++mt)
#pragma unroll
            for (int nt = 0; nt < 4; ++nt)
                acc[mt][nt] = __builtin_amdgcn_wmma_f32_16x16x32_f16(
                    /*neg_a=*/false, af[mt],
                    /*neg_b=*/false, bf[nt],
                    /*c_mod=*/(short)0, acc[mt][nt],
                    /*reuse_a=*/false, /*reuse_b=*/false);

        __syncthreads();   // done reading buf before it is re-filled next iter
    }

    // C layout: VGPR r, lane l -> M = r + 8*(l>>4), N = l&15
#pragma unroll
    for (int nt = 0; nt < 4; ++nt) {
        const int col = tileN + nt * 16 + lrow;
        const float bv = bias[col];
#pragma unroll
        for (int mt = 0; mt < 2; ++mt) {
            const int rbase = tileM + mt * 16 + lhalf * 8;
#pragma unroll
            for (int r = 0; r < 8; ++r)
                C[(size_t)(rbase + r) * N + col] = acc[mt][nt][r] + bv;
        }
    }
}

// ---------------- tail: avg_spikes = 0, mem_out = 0 ----------------
__global__ void write_tail(float* __restrict__ out, size_t off) {
    if (threadIdx.x == 0) { out[off] = 0.0f; out[off + 1] = 0.0f; }
}

// ---------------------------------------------------------------------------
extern "C" void kernel_launch(void* const* d_in, const int* in_sizes, int n_in,
                              void* d_out, int out_size, void* d_ws, size_t ws_size,
                              hipStream_t stream) {
    (void)in_sizes; (void)n_in; (void)out_size; (void)ws_size;

    const int*   ids = (const int*)  d_in[0];
    const float* emb = (const float*)d_in[1];
    const float* W1  = (const float*)d_in[2];
    const float* b1  = (const float*)d_in[3];
    const float* W2  = (const float*)d_in[4];
    const float* b2  = (const float*)d_in[5];
    const float* thr = (const float*)d_in[6];
    float* out = (float*)d_out;

    // workspace carve-up (256B aligned)
    char* ws = (char*)d_ws;
    size_t off = 0;
    auto alloc = [&](size_t bytes) { char* p = ws + off;
                                     off += (bytes + 255) & ~(size_t)255; return p; };
    _Float16* xh   = (_Float16*)alloc((size_t)MTOT  * DMODEL * 2);   //  2 MB
    _Float16* W1h  = (_Float16*)alloc((size_t)HIDDEN* DMODEL * 2);   //  1 MB
    _Float16* W2h  = (_Float16*)alloc((size_t)VOCAB * HIDDEN * 2);   // 65.5 MB
    float*    u    = (float*)   alloc((size_t)MTOT  * HIDDEN * 4);   //  8.4 MB
    _Float16* spk  = (_Float16*)alloc((size_t)MTOT  * HIDDEN * 2);   //  4.2 MB

    // 1) weight casts
    {
        size_t n1 = (size_t)HIDDEN * DMODEL;
        cast_f32_to_f16<<<(unsigned)((n1 + 255) / 256), 256, 0, stream>>>(W1, W1h, n1);
        size_t n2 = (size_t)VOCAB * HIDDEN;
        cast_f32_to_f16<<<(unsigned)((n2 + 255) / 256), 256, 0, stream>>>(W2, W2h, n2);
    }
    // 2) embedding gather
    {
        size_t n = (size_t)MTOT * DMODEL;
        gather_cast<<<(unsigned)((n + 255) / 256), 256, 0, stream>>>(ids, emb, xh);
    }
    // 3) FC1: u[2048,1024] = xh[2048,512] * W1h[1024,512]^T + b1
    {
        dim3 grid(HIDDEN / 64, MTOT / 128);   // (16,16)
        gemm_f16_wmma<<<grid, 128, 0, stream>>>(xh, W1h, b1, u, MTOT, HIDDEN, DMODEL);
    }
    // 4) membrane scan -> spikes (f16, exactly {0,1})
    snn_scan<<<(BB * HIDDEN) / 256, 256, 0, stream>>>(u, thr, spk);

    // 5) FC2: logits[2048,32000] = spk[2048,1024] * W2h[32000,1024]^T + b2
    {
        dim3 grid(VOCAB / 64, MTOT / 128);    // (500,16)
        gemm_f16_wmma<<<grid, 128, 0, stream>>>(spk, W2h, b2, out, MTOT, VOCAB, HIDDEN);
    }
    // 6) scalar tail outputs
    write_tail<<<1, 32, 0, stream>>>(out, (size_t)MTOT * VOCAB);
}
